// MWTF_29016799052177
// MI455X (gfx1250) — matrix operand order repaired and checked by reading, not compiled, not run on hardware
//
#include <hip/hip_runtime.h>
#include <hip/hip_bf16.h>

// ---------------------------------------------------------------------------
// Fused windowed dual-axis attention for MI455X (gfx1250, wave32, WMMA).
// One 8-wave workgroup per (batch, window) block; 256 blocks total.
// All matmuls via v_wmma_f32_16x16x32_f16 (f16 operands, f32 accumulate).
// ---------------------------------------------------------------------------

typedef _Float16 v16h __attribute__((ext_vector_type(16)));
typedef _Float16 v8h  __attribute__((ext_vector_type(8)));
typedef float    v8f  __attribute__((ext_vector_type(8)));

#define WMMA_F32_F16(a, b, c) \
  __builtin_amdgcn_wmma_f32_16x16x32_f16(false, (a), false, (b), (short)0, (c), false, false)

__device__ __forceinline__ v8f zero8() {
  v8f z;
#pragma unroll
  for (int r = 0; r < 8; ++r) z[r] = 0.0f;
  return z;
}

// gfx1250 has a hardware tanh transcendental; use it if the builtin exists,
// otherwise a branch-free fast-exp formulation (avoids libm's exec-divergent path).
__device__ __forceinline__ float fast_tanh(float x) {
#if __has_builtin(__builtin_amdgcn_tanhf)
  return __builtin_amdgcn_tanhf(x);
#elif __has_builtin(__builtin_amdgcn_tanh_f32)
  return __builtin_amdgcn_tanh_f32(x);
#else
  const float xc = fminf(fmaxf(x, -10.0f), 10.0f);
  const float e = __expf(2.0f * xc);
  return (e - 1.0f) / (e + 1.0f);
#endif
}

// A fragment (16x32 f16): lane holds row m = lane&15; k = kh*16 + 0..15
// from a row-major [M][K...] tile with row stride `stride` halves.
__device__ __forceinline__ v16h frag_a(const _Float16* base, int stride, int lane) {
  const int r = lane & 15, kh = lane >> 4;
  const v8h lo = *(const v8h*)(base + r * stride + kh * 16);
  const v8h hi = *(const v8h*)(base + r * stride + kh * 16 + 8);
  v16h a;
#pragma unroll
  for (int i = 0; i < 8; ++i) { a[i] = lo[i]; a[i + 8] = hi[i]; }
  return a;
}

// B fragment (32x16 f16): lane holds column n = lane&15; k = kh*16 + 0..15.
// `p` must point at the lane's 16 contiguous halves (caller adds kh*16).
__device__ __forceinline__ v16h frag_b(const _Float16* p) {
  const v8h lo = *(const v8h*)(p);
  const v8h hi = *(const v8h*)(p + 8);
  v16h b;
#pragma unroll
  for (int i = 0; i < 8; ++i) { b[i] = lo[i]; b[i + 8] = hi[i]; }
  return b;
}

struct SMem {
  alignas(16) _Float16 nrm[32][256];    // layer-normed input      16 KB
  alignas(16) _Float16 qm[32][512];     // Q * (1/sqrt(H))         32 KB
  alignas(16) _Float16 qt[512][32];     // Q^T * (1/sqrt(W))       32 KB
  alignas(16) _Float16 km[32][512];     // K                       32 KB
  alignas(16) _Float16 kt[512][32];     // K^T                     32 KB
  alignas(16) _Float16 vm[32][512];     // V                       32 KB
  alignas(16) _Float16 tbuf[32 * 512];  // t1 [32][512] / featT [512][32]  32 KB
  alignas(16) float    bt[32][32];      // beta_time                4 KB
  alignas(16) _Float16 at[32][32];      // A_time (softmaxed)       2 KB
  alignas(16) _Float16 pst[8][32][16];  // per-wave P staging (B-frag order)  8 KB
};                                      // total ~222 KB (of 320 KB/WGP)

// C = act(A[32xK] @ W^T-stored[NxK] + bias), 64 output tiles over 8 waves.
// act: 0 = linear, 1 = tanh, 2 = relu. Optionally dual-store row-major
// (stride 512, scaled by sR) and transposed (stride 32, scaled by sT).
__device__ __forceinline__ void gemm_block(
    const _Float16* A, int As, const _Float16* W, int Ws, const float* bias,
    int kchunks, int act, float sR, float sT,
    _Float16* outR, _Float16* outT, int lane, int wave) {
  const int n = lane & 15, kh = lane >> 4;
#pragma unroll 1
  for (int i = 0; i < 8; ++i) {
    const int tile = wave * 8 + i;
    const int mt = tile & 1, nt = tile >> 1;
    const int col = nt * 16 + n;
    v8f acc = zero8();
#pragma unroll 2   // two k-steps in flight: overlap b128 loads with wmma
    for (int kc = 0; kc < kchunks; ++kc) {
      v16h a = frag_a(A + (mt * 16) * As + kc * 32, As, lane);
      v16h b = frag_b(W + (size_t)col * Ws + kc * 32 + kh * 16);
      acc = WMMA_F32_F16(a, b, acc);
    }
    const float bv = bias[col];
#pragma unroll
    for (int r = 0; r < 8; ++r) {
      const int m = mt * 16 + kh * 8 + r;
      float v = acc[r] + bv;
      if (act == 1) v = fast_tanh(v);
      if (act == 2) v = fmaxf(v, 0.0f);
      if (outR) outR[m * 512 + col] = (_Float16)(v * sR);
      if (outT) outT[col * 32 + m] = (_Float16)(v * sT);
    }
  }
}

__global__ __launch_bounds__(256) void MWTF_fused_kernel(
    const float* __restrict__ x, const float* __restrict__ gamma,
    const float* __restrict__ beta,
    const _Float16* __restrict__ w1T, const float* __restrict__ b1,
    const _Float16* __restrict__ w2T, const float* __restrict__ b2,
    const _Float16* __restrict__ w3T, const float* __restrict__ b3,
    const _Float16* __restrict__ w4T, const float* __restrict__ b4,
    const _Float16* __restrict__ w5T, const float* __restrict__ b5,
    const _Float16* __restrict__ w6T, const float* __restrict__ b6,
    float* __restrict__ out) {
  __shared__ SMem s;
  const int tid = threadIdx.x;
  const int lane = tid & 31;
  const int wave = tid >> 5;
  const int blk = blockIdx.x;  // (b * 64 + window_idx) -> rows blk*32 .. +31

  // ---- Stage 0: LayerNorm (8 threads per row, 32 elems each) ----
  {
    const int r = tid >> 3, part = tid & 7;
    const float* xr = x + ((size_t)blk * 32 + r) * 256 + part * 32;
    float vals[32];
    float ssum = 0.0f, ssq = 0.0f;
#pragma unroll
    for (int j = 0; j < 32; ++j) {
      float v = xr[j];
      vals[j] = v; ssum += v; ssq += v * v;
    }
#pragma unroll
    for (int off = 1; off < 8; off <<= 1) {
      ssum += __shfl_xor(ssum, off, 32);
      ssq  += __shfl_xor(ssq, off, 32);
    }
    const float mean = ssum * (1.0f / 256.0f);
    const float var  = ssq * (1.0f / 256.0f) - mean * mean;
    const float rstd = rsqrtf(var + 1e-5f);
#pragma unroll
    for (int j = 0; j < 32; ++j) {
      const int c = part * 32 + j;
      s.nrm[r][c] = (_Float16)((vals[j] - mean) * rstd * gamma[c] + beta[c]);
    }
  }
  __syncthreads();

  const float invSH = 0.044194173824159216f;  // 1/sqrt(512)
  const float invSW = 0.17677669529663687f;   // 1/sqrt(32)

  // ---- Stage 1: Q/K/V projection pipelines (two GEMMs each) ----
  // Q: fold 1/sqrt(H) into row-major copy, 1/sqrt(W) into transposed copy.
  gemm_block(&s.nrm[0][0], 256, w1T, 256, b1, 8, 1, 1.0f, 0.0f, s.tbuf, nullptr, lane, wave);
  __syncthreads();
  gemm_block(s.tbuf, 512, w2T, 512, b2, 16, 0, invSH, invSW, &s.qm[0][0], &s.qt[0][0], lane, wave);
  __syncthreads();
  gemm_block(&s.nrm[0][0], 256, w3T, 256, b3, 8, 1, 1.0f, 0.0f, s.tbuf, nullptr, lane, wave);
  __syncthreads();
  gemm_block(s.tbuf, 512, w4T, 512, b4, 16, 0, 1.0f, 1.0f, &s.km[0][0], &s.kt[0][0], lane, wave);
  __syncthreads();
  gemm_block(&s.nrm[0][0], 256, w5T, 256, b5, 8, 2, 1.0f, 0.0f, s.tbuf, nullptr, lane, wave);
  __syncthreads();
  gemm_block(s.tbuf, 512, w6T, 512, b6, 16, 0, 1.0f, 0.0f, &s.vm[0][0], nullptr, lane, wave);
  __syncthreads();

  // ---- Stage 2: beta_time = (Q/sqrtH) @ K^T   [32x32], waves 0..3 ----
  if (wave < 4) {
    const int mt = wave >> 1, nt = wave & 1;
    const int n = lane & 15, kh = lane >> 4;
    const int col = nt * 16 + n;
    v8f acc = zero8();
#pragma unroll 2
    for (int kc = 0; kc < 16; ++kc) {
      v16h a = frag_a(&s.qm[mt * 16][kc * 32], 512, lane);
      v16h b = frag_b(&s.km[col][kc * 32 + kh * 16]);  // B[k=h][n=v] = K[v][h]
      acc = WMMA_F32_F16(a, b, acc);
    }
#pragma unroll
    for (int r = 0; r < 8; ++r) s.bt[mt * 16 + kh * 8 + r][col] = acc[r];
  }
  __syncthreads();

  // ---- Stage 3: A_time = row-softmax(beta_time), one thread per row ----
  if (tid < 32) {
    float row[32];
    float mx = -3.0e38f;
#pragma unroll
    for (int v = 0; v < 32; ++v) { row[v] = s.bt[tid][v]; mx = fmaxf(mx, row[v]); }
    float sm = 0.0f;
#pragma unroll
    for (int v = 0; v < 32; ++v) { row[v] = __expf(row[v] - mx); sm += row[v]; }
    const float inv = 1.0f / sm;
#pragma unroll
    for (int v = 0; v < 32; ++v) s.at[tid][v] = (_Float16)(row[v] * inv);
  }
  __syncthreads();

  // ---- Stage 4: feat = V @ softmax_h(Q^T K / sqrtW), streamed per 16-col tile ----
  _Float16* featT = s.tbuf;  // reuse as [512 g][32 w]
  {
    const int n = lane & 15, kh = lane >> 4;
#pragma unroll 1
    for (int g = 0; g < 4; ++g) {
      const int gt = wave * 4 + g;
      // K^T B-fragment is identical for all 64 wmmas of this column tile: hoist.
      const v16h bK = frag_b(&s.kt[gt * 16 + n][kh * 16]);  // B[w][g] = K^T row
      // Sweep 1: per-column online max & exp-sum over all 512 h.
      float mrun = -3.0e38f, srun = 0.0f;
#pragma unroll 2
      for (int ht = 0; ht < 32; ++ht) {
        v16h a = frag_a(&s.qt[ht * 16][0], 32, lane);       // A[h][w] = Q^T
        v8f d = WMMA_F32_F16(a, bK, zero8());
        float tmax = d[0];
#pragma unroll
        for (int r = 1; r < 8; ++r) tmax = fmaxf(tmax, d[r]);
        tmax = fmaxf(tmax, __shfl_xor(tmax, 16, 32));       // lane-pair shares column
        const float mnew = fmaxf(mrun, tmax);
        float p = 0.0f;
#pragma unroll
        for (int r = 0; r < 8; ++r) p += __expf(d[r] - mnew);
        p += __shfl_xor(p, 16, 32);
        srun = srun * __expf(mrun - mnew) + p;
        mrun = mnew;
      }
      const float invs = 1.0f / srun;
      // Sweep 2: recompute, normalize, stage P in B-fragment order, GEMM V@P.
      v8f acc0 = zero8(), acc1 = zero8();
#pragma unroll 1
      for (int hb = 0; hb < 16; ++hb) {
#pragma unroll
        for (int ts = 0; ts < 2; ++ts) {
          const int ht = hb * 2 + ts;
          v16h a = frag_a(&s.qt[ht * 16][0], 32, lane);
          v8f d = WMMA_F32_F16(a, bK, zero8());
#pragma unroll
          for (int r = 0; r < 8; ++r) {
            const int k = ts * 16 + kh * 8 + r;  // h within 32-chunk
            s.pst[wave][(k >> 4) * 16 + n][k & 15] =
                (_Float16)(__expf(d[r] - mrun) * invs);
          }
        }
        // Same-wave LDS ops are in-order; compiler inserts dscnt waits.
        v16h bP = frag_b(&s.pst[wave][lane][0]);
        v16h a0 = frag_a(&s.vm[0][hb * 32], 512, lane);
        v16h a1 = frag_a(&s.vm[16][hb * 32], 512, lane);
        acc0 = WMMA_F32_F16(a0, bP, acc0);
        acc1 = WMMA_F32_F16(a1, bP, acc1);
      }
#pragma unroll
      for (int r = 0; r < 8; ++r) {
        const int m = kh * 8 + r;
        featT[(gt * 16 + n) * 32 + m]      = (_Float16)acc0[r];
        featT[(gt * 16 + n) * 32 + 16 + m] = (_Float16)acc1[r];
      }
    }
  }
  __syncthreads();

  // ---- Stage 5: out = A_time @ feat  [32x512] -> global ----
  {
    const int n = lane & 15, kh = lane >> 4;
    float* outBase = out + (size_t)blk * 32 * 512;
#pragma unroll 1
    for (int i = 0; i < 8; ++i) {
      const int tile = wave * 8 + i;
      const int mt = tile & 1, nt = tile >> 1;
      const int col = nt * 16 + n;
      v16h a = frag_a(&s.at[mt * 16][0], 32, lane);
      v16h b = frag_b(&featT[col * 32 + kh * 16]);   // B[v][g] = featT row g
      v8f d = WMMA_F32_F16(a, b, zero8());
#pragma unroll
      for (int r = 0; r < 8; ++r)
        outBase[(size_t)(mt * 16 + kh * 8 + r) * 512 + col] = d[r];
    }
  }
}

// Convert fp32 weight [K][N] -> f16 transposed [N][K] in workspace.
__global__ void MWTF_wprep_kernel(const float* __restrict__ src,
                                  _Float16* __restrict__ dst, int K, int N) {
  const int idx = blockIdx.x * blockDim.x + threadIdx.x;
  if (idx < K * N) {
    const int n = idx / K;
    const int k = idx - n * K;
    dst[idx] = (_Float16)src[(size_t)k * N + n];
  }
}

extern "C" void kernel_launch(void* const* d_in, const int* in_sizes, int n_in,
                              void* d_out, int out_size, void* d_ws, size_t ws_size,
                              hipStream_t stream) {
  (void)in_sizes; (void)n_in; (void)out_size; (void)ws_size;
  const float* feature = (const float*)d_in[0];
  const float* gamma   = (const float*)d_in[1];
  const float* beta    = (const float*)d_in[2];
  const float* w1 = (const float*)d_in[3];  const float* b1 = (const float*)d_in[4];
  const float* w2 = (const float*)d_in[5];  const float* b2 = (const float*)d_in[6];
  const float* w3 = (const float*)d_in[7];  const float* b3 = (const float*)d_in[8];
  const float* w4 = (const float*)d_in[9];  const float* b4 = (const float*)d_in[10];
  const float* w5 = (const float*)d_in[11]; const float* b5 = (const float*)d_in[12];
  const float* w6 = (const float*)d_in[13]; const float* b6 = (const float*)d_in[14];
  float* out = (float*)d_out;

  _Float16* ws = (_Float16*)d_ws;
  const size_t S1 = 256 * 512, S2 = 512 * 512;
  _Float16* w1T = ws;
  _Float16* w2T = w1T + S1;
  _Float16* w3T = w2T + S2;
  _Float16* w4T = w3T + S1;
  _Float16* w5T = w4T + S2;
  _Float16* w6T = w5T + S1;  // total 1,179,648 halves = 2.25 MB of workspace

  const int thr = 256;
  MWTF_wprep_kernel<<<(int)((S1 + thr - 1) / thr), thr, 0, stream>>>(w1, w1T, 256, 512);
  MWTF_wprep_kernel<<<(int)((S2 + thr - 1) / thr), thr, 0, stream>>>(w2, w2T, 512, 512);
  MWTF_wprep_kernel<<<(int)((S1 + thr - 1) / thr), thr, 0, stream>>>(w3, w3T, 256, 512);
  MWTF_wprep_kernel<<<(int)((S2 + thr - 1) / thr), thr, 0, stream>>>(w4, w4T, 512, 512);
  MWTF_wprep_kernel<<<(int)((S1 + thr - 1) / thr), thr, 0, stream>>>(w5, w5T, 256, 512);
  MWTF_wprep_kernel<<<(int)((S2 + thr - 1) / thr), thr, 0, stream>>>(w6, w6T, 512, 512);

  // 4 batches * 64 windows = 256 blocks of [32, 256]
  MWTF_fused_kernel<<<256, 256, 0, stream>>>(
      feature, gamma, beta, w1T, b1, w2T, b2, w3T, b3, w4T, b4, w5T, b5, w6T, b6, out);
}
